// VanillaRNN_51531017617809
// MI455X (gfx1250) — compile-verified, matrix operand order checked by reading, not run on hardware
//
#include <hip/hip_runtime.h>
#include <hip/hip_bf16.h>

#define B_ 256
#define T_ 128
#define H_ 2048
#define C_ 10
#define NBLK_ 64   // persistent blocks; each owns a 32-column slice of W

#if __has_builtin(__builtin_amdgcn_tanhf)
#define TANH_(v) __builtin_amdgcn_tanhf(v)
#else
#define TANH_(v) tanhf(v)
#endif

typedef __attribute__((ext_vector_type(16))) __bf16 v16bf;
typedef __attribute__((ext_vector_type(8)))  __bf16 v8bf;
typedef __attribute__((ext_vector_type(8)))  float  v8f;

// WMMA bf16 A fragment (16 rows x 32 K) from global, row stride H_.
// Per ISA layout: lanes 0-15 hold K=0..7 / K=16..23 of row (lane&15);
// lanes 16-31 hold K=8..15 / K=24..31.
__device__ __forceinline__ v16bf load_frag_g(const __hip_bfloat16* base,
                                             int row0, int k0, int lane) {
  int r  = row0 + (lane & 15);
  int kb = k0 + ((lane >> 4) << 3);
  const __bf16* p = reinterpret_cast<const __bf16*>(base) + r * H_ + kb;
  union { v16bf v; v8bf h[2]; } u;
  u.h[0] = *reinterpret_cast<const v8bf*>(p);
  u.h[1] = *reinterpret_cast<const v8bf*>(p + 16);
  return u.v;
}

// WMMA bf16 B fragment from LDS, tiled layout [chunk][32 rows][32 k]:
// element (row r, k = chunk*32 + kk) at index chunk*1024 + r*32 + kk.
// Within one chunk both tiles' fragments are at small immediate offsets.
__device__ __forceinline__ v16bf load_frag_l(const __hip_bfloat16* lds,
                                             int row0, int chunkBase, int lane) {
  int idx = chunkBase + (row0 + (lane & 15)) * 32 + ((lane >> 4) << 3);
  const __bf16* p = reinterpret_cast<const __bf16*>(lds) + idx;
  union { v16bf v; v8bf h[2]; } u;
  u.h[0] = *reinterpret_cast<const v8bf*>(p);
  u.h[1] = *reinterpret_cast<const v8bf*>(p + 16);
  return u.v;
}

// Grid-wide generation barrier (all NBLK_ blocks resident).
__device__ __forceinline__ void grid_barrier(unsigned* cnt, unsigned* gen) {
  __threadfence();          // release this thread's stores to device scope
  __syncthreads();
  if (threadIdx.x == 0) {
    unsigned g = __hip_atomic_load(gen, __ATOMIC_RELAXED, __HIP_MEMORY_SCOPE_AGENT);
    unsigned a = __hip_atomic_fetch_add(cnt, 1u, __ATOMIC_ACQ_REL, __HIP_MEMORY_SCOPE_AGENT);
    if (a == NBLK_ - 1) {
      __hip_atomic_store(cnt, 0u, __ATOMIC_RELAXED, __HIP_MEMORY_SCOPE_AGENT);
      __hip_atomic_fetch_add(gen, 1u, __ATOMIC_RELEASE, __HIP_MEMORY_SCOPE_AGENT);
    } else {
      while (__hip_atomic_load(gen, __ATOMIC_ACQUIRE, __HIP_MEMORY_SCOPE_AGENT) == g)
        __builtin_amdgcn_s_sleep(2);
    }
  }
  __syncthreads();
  __threadfence();          // acquire-side: invalidate stale lines before reads
}

// Persistent RNN: all 128 steps in one launch.
// Grid: 64 blocks x 256 threads (8 waves). Block b owns cols n0=b*32 of W
// (staged in 128KB LDS as bf16, k-chunk tiled); wave w owns rows m0=w*32;
// each wave computes a 2x2 block of 16x16 WMMA tiles per step.
__global__ __launch_bounds__(256) void rnn_persist(
    const float* __restrict__ W,
    const float* __restrict__ x,
    const float* __restrict__ U,
    const float* __restrict__ bh,
    __hip_bfloat16* __restrict__ hb0,
    __hip_bfloat16* __restrict__ hb1,
    float* __restrict__ hlast,
    unsigned* __restrict__ bar)
{
  extern __shared__ unsigned short WldsRaw[];          // 32 * 2048 bf16 = 128KB
  __hip_bfloat16* Wlds = reinterpret_cast<__hip_bfloat16*>(WldsRaw);

  const int tid  = threadIdx.x;
  const int lane = tid & 31;
  const int wave = tid >> 5;
  const int n0   = blockIdx.x * 32;   // global output-column base of this block
  const int m0   = wave * 32;         // batch-row base of this wave

  // Stage this block's W slice into LDS (fp32 -> bf16), rows n0..n0+31,
  // tiled as [k/32][row][k%32].
  for (int i = tid; i < 32 * H_; i += 256) {
    int r = i >> 11, k = i & (H_ - 1);
    int dst = ((k >> 5) << 10) | (r << 5) | (k & 31);
    Wlds[dst] = __float2bfloat16(W[(size_t)(n0 + r) * H_ + k]);
  }
  // Cooperatively zero h0 (bf16).
  {
    const int share = (B_ * H_) / NBLK_;
    __hip_bfloat16* dst = hb0 + (size_t)blockIdx.x * share;
    for (int i = tid; i < share; i += 256) dst[i] = __float2bfloat16(0.0f);
  }
  grid_barrier(bar, bar + 1);

  // Per-wave constants for the two n-tiles (fixed across all steps).
  const int lhalf = lane >> 4;
  const int ncol  = lane & 15;
  float un[2], bn[2];
#pragma unroll
  for (int j = 0; j < 2; ++j) {
    const int n = n0 + j * 16 + ncol;
    un[j] = U[n];
    bn[j] = bh[n];
  }

  for (int t = 0; t < T_; ++t) {
    const __hip_bfloat16* hp = (t & 1) ? hb1 : hb0;
    __hip_bfloat16*       hn = (t & 1) ? hb0 : hb1;

    v8f acc[2][2] = {};
#pragma unroll 2
    for (int c = 0; c < H_ / 32; ++c) {
      const int k0 = c * 32;
      const int cb = c * 1024;                         // LDS chunk base (elems)
      v16bf a0 = load_frag_g(hp,   m0,      k0, lane); // A from global (L2-hot)
      v16bf a1 = load_frag_g(hp,   m0 + 16, k0, lane);
      v16bf b0 = load_frag_l(Wlds, 0,       cb, lane); // B from LDS
      v16bf b1 = load_frag_l(Wlds, 16,      cb, lane);
      acc[0][0] = __builtin_amdgcn_wmma_f32_16x16x32_bf16(false, a0, false, b0,
                      (short)0, acc[0][0], false, false);
      acc[0][1] = __builtin_amdgcn_wmma_f32_16x16x32_bf16(false, a0, false, b1,
                      (short)0, acc[0][1], false, false);
      acc[1][0] = __builtin_amdgcn_wmma_f32_16x16x32_bf16(false, a1, false, b0,
                      (short)0, acc[1][0], false, false);
      acc[1][1] = __builtin_amdgcn_wmma_f32_16x16x32_bf16(false, a1, false, b1,
                      (short)0, acc[1][1], false, false);
    }

    // Epilogue: h_new = tanh(acc + x[:,t]*U + bh).
    // C/D layout: lane -> col n0 + j*16 + (lane&15); VGPR r -> row r + 8*lhalf.
    float xv[2][8];
#pragma unroll
    for (int i = 0; i < 2; ++i)
#pragma unroll
      for (int r = 0; r < 8; ++r)
        xv[i][r] = x[(m0 + i * 16 + r + 8 * lhalf) * T_ + t];

#pragma unroll
    for (int j = 0; j < 2; ++j) {
      const int n = n0 + j * 16 + ncol;
#pragma unroll
      for (int i = 0; i < 2; ++i) {
#pragma unroll
        for (int r = 0; r < 8; ++r) {
          const int m = m0 + i * 16 + r + 8 * lhalf;
          const float v = TANH_(acc[i][j][r] + xv[i][r] * un[j] + bn[j]);
          hn[m * H_ + n] = __float2bfloat16(v);
          if (t == T_ - 1) hlast[m * H_ + n] = v;
        }
      }
    }

    if (t != T_ - 1) grid_barrier(bar, bar + 1);
  }
}

// out[b][c] = dot(h_last[b,:], V[c,:]) + bp[c]; one wave per (b,c).
__global__ __launch_bounds__(256) void rnn_out(
    const float* __restrict__ h, const float* __restrict__ V,
    const float* __restrict__ bp, float* __restrict__ out) {
  const int lane = threadIdx.x & 31;
  const int wid  = (blockIdx.x * blockDim.x + threadIdx.x) >> 5;
  if (wid >= B_ * C_) return;
  const int b = wid / C_, c = wid % C_;
  float s = 0.0f;
  for (int k = lane; k < H_; k += 32)
    s += h[b * H_ + k] * V[c * H_ + k];
#pragma unroll
  for (int off = 16; off > 0; off >>= 1)
    s += __shfl_xor(s, off, 32);
  if (lane == 0) out[b * C_ + c] = s + bp[c];
}

extern "C" void kernel_launch(void* const* d_in, const int* in_sizes, int n_in,
                              void* d_out, int out_size, void* d_ws, size_t ws_size,
                              hipStream_t stream) {
  const float* x  = (const float*)d_in[0];  // [B,T]
  const float* U  = (const float*)d_in[1];  // [H,1]
  const float* W  = (const float*)d_in[2];  // [H,H]
  const float* V  = (const float*)d_in[3];  // [C,H]
  const float* bh = (const float*)d_in[4];  // [H]
  const float* bp = (const float*)d_in[5];  // [C]

  float* hlast = (float*)d_out;             // [B,H] fp32
  float* outp  = hlast + (size_t)B_ * H_;   // [B,C] fp32

  char* ws = (char*)d_ws;
  unsigned* bar = (unsigned*)ws;                                  // 2 x u32
  __hip_bfloat16* hb0 = (__hip_bfloat16*)(ws + 256);              // 1 MB
  __hip_bfloat16* hb1 = hb0 + (size_t)B_ * H_;                    // 1 MB

  hipMemsetAsync(bar, 0, 2 * sizeof(unsigned), stream);

  const size_t lds_bytes = (size_t)32 * H_ * sizeof(unsigned short); // 128 KB
  rnn_persist<<<NBLK_, 256, lds_bytes, stream>>>(W, x, U, bh, hb0, hb1,
                                                 hlast, bar);

  rnn_out<<<(B_ * C_ * 32 + 255) / 256, 256, 0, stream>>>(hlast, V, bp, outp);
}
